// MultiHeadedSelfAttention_9302899163706
// MI455X (gfx1250) — matrix operand order repaired and checked by reading, not compiled, not run on hardware
//
#include <hip/hip_runtime.h>

#define Bdim 2
#define Sdim 4096
#define Ddim 768
#define Hdim 12
#define HDdim 64

typedef __attribute__((ext_vector_type(16))) _Float16 v16h;
typedef __attribute__((ext_vector_type(8)))  _Float16 v8h;
typedef __attribute__((ext_vector_type(8)))  float    v8f;
typedef __attribute__((ext_vector_type(4)))  float    f4;

union V16U { v16h v; v8h h[2]; };

__device__ __forceinline__ v8f wmma_f16(v16h a, v16h b, v8f c) {
  // D = A(16x32 f16) * B(32x16 f16) + C(16x16 f32)
  return __builtin_amdgcn_wmma_f32_16x16x32_f16(false, a, false, b, (short)0, c,
                                                false, false);
}

__device__ __forceinline__ float lane_max16(float x) {
  x = fmaxf(x, __shfl_xor(x, 1, 32));
  x = fmaxf(x, __shfl_xor(x, 2, 32));
  x = fmaxf(x, __shfl_xor(x, 4, 32));
  x = fmaxf(x, __shfl_xor(x, 8, 32));
  return x;
}
__device__ __forceinline__ float lane_sum16(float x) {
  x += __shfl_xor(x, 1, 32);
  x += __shfl_xor(x, 2, 32);
  x += __shfl_xor(x, 4, 32);
  x += __shfl_xor(x, 8, 32);
  return x;
}

// ---------------------------------------------------------------------------
// Kernel 0: one-shot fp32 -> f16 downconvert (x and the three weight matrices)
// so the GEMM kernel does pure f16 loads with no per-tile cvt chains.
// ---------------------------------------------------------------------------
__global__ __launch_bounds__(256) void cvt_f16_kernel(
    const float* __restrict__ src, _Float16* __restrict__ dst, int n8)
{
  const int i = blockIdx.x * 256 + threadIdx.x;
  if (i < n8) {
    f4 a = *(const f4*)(src + (size_t)i * 8);
    f4 b = *(const f4*)(src + (size_t)i * 8 + 4);
    v8h o;
    #pragma unroll
    for (int j = 0; j < 4; ++j) {
      o[j]     = (_Float16)a[j];
      o[4 + j] = (_Float16)b[j];
    }
    *(v8h*)(dst + (size_t)i * 8) = o;
  }
}

// ---------------------------------------------------------------------------
// Kernel 1: fused QKV projection, pure-f16 operands.  out = x @ W^T + b.
//   z==0 -> Q row-major (pre-scaled by 1/8 = 1/sqrt(HD))
//   z==1 -> K row-major
//   z==2 -> V transposed: vt[((b*H+h)*64+hd)*S + s]
// One wave computes a 32x64 tile (2 M-subtiles share each B fragment).
// ---------------------------------------------------------------------------
__global__ __launch_bounds__(128) void qkv_proj_kernel(
    const _Float16* __restrict__ x16,
    const _Float16* __restrict__ wq16, const float* __restrict__ bq,
    const _Float16* __restrict__ wk16, const float* __restrict__ bk,
    const _Float16* __restrict__ wv16, const float* __restrict__ bv,
    _Float16* __restrict__ qbuf, _Float16* __restrict__ kbuf,
    _Float16* __restrict__ vtbuf)
{
  __shared__ _Float16 sh[4 * 32 * 64];  // per-wave 32x64 f16 transpose buffer
  const int lane = threadIdx.x & 31;
  const int wid  = threadIdx.x >> 5;
  const int z    = blockIdx.z;
  const _Float16* W = (z == 0) ? wq16 : (z == 1) ? wk16 : wv16;
  const float* bias = (z == 0) ? bq : (z == 1) ? bk : bv;
  const int m0 = (blockIdx.x * 4 + wid) * 32;   // global row (b*S+s)
  const int n0 = blockIdx.y * 64;               // output feature base
  const int nlan = lane & 15;
  const int half = lane >> 4;
  const int akoff = half ? 16 : 0;   // A frag: lanes>=16 hold K 16-31
  const int bkoff = half ? 16 : 0;   // B frag: lanes>=16 hold K 16-31

  v8f c[2][4];
  #pragma unroll
  for (int t = 0; t < 4; ++t) {
    const float bv0 = bias[n0 + t * 16 + nlan];   // C[m][n] init = bias[n]
    #pragma unroll
    for (int r = 0; r < 8; ++r) { c[0][t][r] = bv0; c[1][t][r] = bv0; }
  }

  for (int k0 = 0; k0 < Ddim; k0 += 32) {
    V16U a0, a1;
    {
      const _Float16* p0 = x16 + (size_t)(m0 + nlan) * Ddim + k0 + akoff;
      const _Float16* p1 = p0 + 16 * Ddim;
      a0.h[0] = *(const v8h*)(p0);
      a0.h[1] = *(const v8h*)(p0 + 8);
      a1.h[0] = *(const v8h*)(p1);
      a1.h[1] = *(const v8h*)(p1 + 8);
    }
    #pragma unroll
    for (int t = 0; t < 4; ++t) {
      // B operand = W^T tile: lane n holds row (n0+t*16+n) of W, contiguous K
      v16h bf = *(const v16h*)(W + (size_t)(n0 + t * 16 + nlan) * Ddim + k0 + bkoff);
      c[0][t] = wmma_f16(a0.v, bf, c[0][t]);
      c[1][t] = wmma_f16(a1.v, bf, c[1][t]);
    }
  }

  if (z == 2) {
    // V: store transposed f16. Lane holds column n, rows r+half*8 (consecutive s).
    const int b_  = m0 / Sdim;
    const int m0s = m0 % Sdim;
    #pragma unroll
    for (int g = 0; g < 2; ++g)
      #pragma unroll
      for (int t = 0; t < 4; ++t) {
        const int feat = n0 + t * 16 + nlan;
        const int h  = feat >> 6;
        const int hd = feat & 63;
        v8h pk;
        #pragma unroll
        for (int r = 0; r < 8; ++r) pk[r] = (_Float16)c[g][t][r];
        *(v8h*)(vtbuf + ((size_t)(b_ * Hdim + h) * HDdim + hd) * Sdim + m0s +
                g * 16 + half * 8) = pk;
      }
  } else {
    // Q/K: LDS transpose so each lane emits contiguous 16B global stores.
    const float osc = (z == 0) ? 0.125f : 1.0f;  // bake 1/sqrt(64) into Q
    _Float16* s = sh + wid * (32 * 64);
    #pragma unroll
    for (int g = 0; g < 2; ++g)
      #pragma unroll
      for (int t = 0; t < 4; ++t)
        #pragma unroll
        for (int r = 0; r < 8; ++r)
          s[(g * 16 + r + half * 8) * 64 + t * 16 + nlan] =
              (_Float16)(c[g][t][r] * osc);
    asm volatile("s_wait_dscnt 0" ::: "memory");  // wave-local LDS handoff
    _Float16* dstbuf = (z == 0) ? qbuf : kbuf;
    const _Float16* src = s + lane * 64;                       // one row/lane
    _Float16* dst = dstbuf + (size_t)(m0 + lane) * Ddim + n0;  // 128B/lane
    #pragma unroll
    for (int j = 0; j < 8; ++j)
      *(v8h*)(dst + j * 8) = *(const v8h*)(src + j * 8);
  }
}

// ---------------------------------------------------------------------------
// Kernel 2: flash attention. One wave owns a 32-query tile of one (b,h),
// streams 32 keys/iteration: 8 score WMMAs + online softmax + 8 PV WMMAs.
// Every K/V fragment is reused by both 16-query subtiles.
// ---------------------------------------------------------------------------
__global__ __launch_bounds__(128) void attn_kernel(
    const _Float16* __restrict__ Q, const _Float16* __restrict__ K,
    const _Float16* __restrict__ Vt, const int* __restrict__ mask,
    float* __restrict__ out)
{
  __shared__ unsigned char smemraw[4 * 4096];  // 4 KB per wave
  const int lane = threadIdx.x & 31;
  const int wid  = threadIdx.x >> 5;
  const int tile = blockIdx.x * 4 + wid;       // 0 .. B*H*(S/32)-1
  const int qt = tile & 127;                   // S/32 = 128
  const int bh = tile >> 7;
  const int h  = bh % Hdim;
  const int b  = bh / Hdim;
  const int q0 = qt * 32;
  const int nlan = lane & 15;
  const int half = lane >> 4;
  const int akoff = half ? 16 : 0;
  const int bkoff = half ? 16 : 0;

  const _Float16* Qb = Q  + (size_t)b * Sdim * Ddim + h * HDdim;
  const _Float16* Kb = K  + (size_t)b * Sdim * Ddim + h * HDdim;
  const _Float16* Vb = Vt + (size_t)(b * Hdim + h) * HDdim * Sdim;
  const int* mb = mask + b * Sdim;

  // Q A-fragments: 2 query subtiles x 2 HD chunks; Q already carries 1/8 scale
  V16U aq[2][2];
  #pragma unroll
  for (int qi = 0; qi < 2; ++qi) {
    const _Float16* p = Qb + (size_t)(q0 + qi * 16 + nlan) * Ddim + akoff;
    aq[qi][0].h[0] = *(const v8h*)(p);
    aq[qi][0].h[1] = *(const v8h*)(p + 8);
    aq[qi][1].h[0] = *(const v8h*)(p + 32);
    aq[qi][1].h[1] = *(const v8h*)(p + 40);
  }

  float Mr[2][8], Lr[2][8];
  #pragma unroll
  for (int qi = 0; qi < 2; ++qi)
    #pragma unroll
    for (int r = 0; r < 8; ++r) { Mr[qi][r] = -3.0e38f; Lr[qi][r] = 0.0f; }
  v8f acc[2][4] = {};

  _Float16* pbuf = (_Float16*)(smemraw + wid * 4096);
  const float L2E = 1.4426950408889634f;

  for (int k0 = 0; k0 < Sdim; k0 += 32) {
    if (k0 + 32 < Sdim) {  // prefetch next key-block (global_prefetch_b8)
      __builtin_prefetch(Kb + (size_t)(k0 + 32 + nlan) * Ddim + bkoff, 0, 3);
      __builtin_prefetch(Kb + (size_t)(k0 + 48 + nlan) * Ddim + bkoff, 0, 3);
      __builtin_prefetch(Vb + (size_t)lane * Sdim + k0 + 32, 0, 3);
      __builtin_prefetch(Vb + (size_t)(32 + lane) * Sdim + k0 + 32, 0, 3);
    }
    // K^T fragments for two 16-key tiles x two HD chunks (shared by subtiles)
    v16h kb00 = *(const v16h*)(Kb + (size_t)(k0 + nlan) * Ddim + bkoff);
    v16h kb01 = *(const v16h*)(Kb + (size_t)(k0 + nlan) * Ddim + 32 + bkoff);
    v16h kb10 = *(const v16h*)(Kb + (size_t)(k0 + 16 + nlan) * Ddim + bkoff);
    v16h kb11 = *(const v16h*)(Kb + (size_t)(k0 + 16 + nlan) * Ddim + 32 + bkoff);

    const float pen0 = -10000.0f * (1.0f - (float)mb[k0 + nlan]);
    const float pen1 = -10000.0f * (1.0f - (float)mb[k0 + 16 + nlan]);

    #pragma unroll
    for (int qi = 0; qi < 2; ++qi) {
      v8f s0 = {}; v8f s1 = {};
      s0 = wmma_f16(aq[qi][0].v, kb00, s0);
      s0 = wmma_f16(aq[qi][1].v, kb01, s0);
      s1 = wmma_f16(aq[qi][0].v, kb10, s1);
      s1 = wmma_f16(aq[qi][1].v, kb11, s1);

      float alr[8];
      #pragma unroll
      for (int r = 0; r < 8; ++r) {
        float x0 = s0[r] + pen0;
        float x1 = s1[r] + pen1;
        float mx = lane_max16(fmaxf(x0, x1));
        float Mn = fmaxf(Mr[qi][r], mx);
        float al = __builtin_amdgcn_exp2f((Mr[qi][r] - Mn) * L2E);
        float p0 = __builtin_amdgcn_exp2f((x0 - Mn) * L2E);
        float p1 = __builtin_amdgcn_exp2f((x1 - Mn) * L2E);
        Lr[qi][r] = Lr[qi][r] * al + lane_sum16(p0 + p1);
        Mr[qi][r] = Mn;
        alr[r] = al;
        // stash P in LDS (D-layout -> row-major 16x32) for A-operand reload
        pbuf[qi * 512 + (r + half * 8) * 32 + nlan]      = (_Float16)p0;
        pbuf[qi * 512 + (r + half * 8) * 32 + 16 + nlan] = (_Float16)p1;
      }
      #pragma unroll
      for (int t = 0; t < 4; ++t)
        #pragma unroll
        for (int r = 0; r < 8; ++r)
          acc[qi][t][r] *= alr[r];
    }

    asm volatile("s_wait_dscnt 0" ::: "memory");
    V16U ap0, ap1;
    ap0.h[0] = *(const v8h*)(pbuf + nlan * 32 + akoff);
    ap0.h[1] = *(const v8h*)(pbuf + nlan * 32 + 8 + akoff);
    ap1.h[0] = *(const v8h*)(pbuf + 512 + nlan * 32 + akoff);
    ap1.h[1] = *(const v8h*)(pbuf + 512 + nlan * 32 + 8 + akoff);

    // out += P(16x32) * V(32keys x 64hd); each V fragment feeds both subtiles
    #pragma unroll
    for (int t = 0; t < 4; ++t) {
      v16h vb = *(const v16h*)(Vb + (size_t)(t * 16 + nlan) * Sdim + k0 + bkoff);
      acc[0][t] = wmma_f16(ap0.v, vb, acc[0][t]);
      acc[1][t] = wmma_f16(ap1.v, vb, acc[1][t]);
    }
  }

  // epilogue: normalize, stage 16x64 f32 tile in LDS, coalesced 128B/lane stores
  float* so = (float*)pbuf;
  #pragma unroll
  for (int qi = 0; qi < 2; ++qi) {
    float inv[8];
    #pragma unroll
    for (int r = 0; r < 8; ++r) inv[r] = 1.0f / Lr[qi][r];
    #pragma unroll
    for (int t = 0; t < 4; ++t)
      #pragma unroll
      for (int r = 0; r < 8; ++r)
        so[(r + half * 8) * 64 + t * 16 + nlan] = acc[qi][t][r] * inv[r];
    asm volatile("s_wait_dscnt 0" ::: "memory");
    const int row = lane >> 1;
    const int hf2 = lane & 1;
    const float* src = so + row * 64 + hf2 * 32;
    float* dst = out + (size_t)(b * Sdim + q0 + qi * 16 + row) * Ddim +
                 h * HDdim + hf2 * 32;
    #pragma unroll
    for (int j = 0; j < 8; ++j)
      *(f4*)(dst + j * 4) = *(const f4*)(src + j * 4);
  }
}

extern "C" void kernel_launch(void* const* d_in, const int* in_sizes, int n_in,
                              void* d_out, int out_size, void* d_ws, size_t ws_size,
                              hipStream_t stream) {
  const float* x  = (const float*)d_in[0];
  const int* mask = (const int*)d_in[1];
  const float* Wq = (const float*)d_in[2];
  const float* bq = (const float*)d_in[3];
  const float* Wk = (const float*)d_in[4];
  const float* bk = (const float*)d_in[5];
  const float* Wv = (const float*)d_in[6];
  const float* bv = (const float*)d_in[7];
  float* out = (float*)d_out;

  const size_t nx = (size_t)Bdim * Sdim * Ddim;  // 6,291,456
  const size_t nw = (size_t)Ddim * Ddim;         // 589,824
  _Float16* x16   = (_Float16*)d_ws;   // 12.6 MB
  _Float16* wq16  = x16  + nx;         // 1.1 MB
  _Float16* wk16  = wq16 + nw;
  _Float16* wv16  = wk16 + nw;
  _Float16* qbuf  = wv16 + nw;         // 12.6 MB
  _Float16* kbuf  = qbuf + nx;         // 12.6 MB
  _Float16* vtbuf = kbuf + nx;         // 12.6 MB (transposed V)

  // Kernel 0: one-shot downconverts
  cvt_f16_kernel<<<dim3((nx / 8 + 255) / 256), 256, 0, stream>>>(x, x16, nx / 8);
  cvt_f16_kernel<<<dim3((nw / 8 + 255) / 256), 256, 0, stream>>>(Wq, wq16, nw / 8);
  cvt_f16_kernel<<<dim3((nw / 8 + 255) / 256), 256, 0, stream>>>(Wk, wk16, nw / 8);
  cvt_f16_kernel<<<dim3((nw / 8 + 255) / 256), 256, 0, stream>>>(Wv, wv16, nw / 8);

  // Kernel 1: QKV projections (32x64 tile per wave)
  dim3 g1((Bdim * Sdim) / 128, Ddim / 64, 3);   // (64, 12, 3)
  qkv_proj_kernel<<<g1, 128, 0, stream>>>(x16, wq16, bq, wk16, bk, wv16, bv,
                                          qbuf, kbuf, vtbuf);

  // Kernel 2: flash attention (32 queries per wave)
  dim3 g2((Bdim * Hdim * (Sdim / 32)) / 4);     // 768 blocks, 4 waves each
  attn_kernel<<<g2, 128, 0, stream>>>(qbuf, kbuf, vtbuf, mask, out);
}
